// GCN_85804856639970
// MI455X (gfx1250) — compile-verified, matrix operand order checked by reading, not run on hardware
//
#include <hip/hip_runtime.h>
#include <hip/hip_bf16.h>

typedef __attribute__((ext_vector_type(16))) __bf16 v16bf;
typedef __attribute__((ext_vector_type(8)))  float  v8f;

#define D_IN  128
#define D_HID 128
#define D_OUT 64

// ---------------- degree / normalization ----------------

__global__ void init_deg_kernel(float* __restrict__ deg, int n) {
    int i = blockIdx.x * blockDim.x + threadIdx.x;
    if (i < n) deg[i] = 1.0f;                 // self-loop contributes 1
}

__global__ void count_deg_kernel(const int* __restrict__ ei, float* __restrict__ deg, int nE) {
    int e = blockIdx.x * blockDim.x + threadIdx.x;
    if (e < nE) atomicAdd(&deg[ei[nE + e]], 1.0f);   // dst row of edge_index
}

__global__ void rsqrt_kernel(float* __restrict__ deg, int n) {
    int i = blockIdx.x * blockDim.x + threadIdx.x;
    if (i < n) deg[i] = rsqrtf(deg[i]);       // deg >= 1 always
}

__global__ void zero_kernel(float* __restrict__ p, long long n) {
    long long i = (long long)blockIdx.x * blockDim.x + threadIdx.x;
    if (i < n) p[i] = 0.0f;
}

// ---------------- weight pre-pack into WMMA B-fragment order (bf16) ----------------
// Fragment layout (16-bit B 32x16, mirroring the A-matrix K mapping, lane = column):
//   lanes 0-15  (half=0): elems 0..7 -> K=k0+0..7,  elems 8..15 -> K=k0+16..23
//   lanes 16-31 (half=1): elems 0..7 -> K=k0+8..15, elems 8..15 -> K=k0+24..31
// Packed index: (((kt*NT + t)*32 + lane)*16 + elem)  -> each lane reads one
// contiguous 32-byte v16bf per (ktile, coltile).

template <int K, int NOUT>
__global__ __launch_bounds__(256)
void pack_w_kernel(const float* __restrict__ W, __bf16* __restrict__ P) {
    constexpr int NT = NOUT / 16;
    int idx = blockIdx.x * blockDim.x + threadIdx.x;
    if (idx >= K * NOUT) return;
    const int j    = idx & 15;
    const int lane = (idx >> 4) & 31;
    const int ft   = idx >> 9;           // kt*NT + t
    const int kt   = ft / NT;
    const int t    = ft % NT;
    const int half = lane >> 4;
    const int l15  = lane & 15;
    const int jj   = j & 7;
    const int k    = kt * 32 + ((j < 8) ? (half * 8 + jj) : (16 + half * 8 + jj));
    const int col  = t * 16 + l15;
    P[idx] = (__bf16)W[(size_t)k * NOUT + col];
}

// ---------------- bf16 WMMA GEMM: Out = pre(A[M,K]) @ W[K,NOUT] (+bout) ----------------

__device__ __forceinline__ float4 relu_bias4(float4 a, float4 b) {
    a.x = fmaxf(a.x + b.x, 0.0f);
    a.y = fmaxf(a.y + b.y, 0.0f);
    a.z = fmaxf(a.z + b.z, 0.0f);
    a.w = fmaxf(a.w + b.w, 0.0f);
    return a;
}

template <int K, int NOUT, bool FUSE_IN, bool BIAS_OUT>
__global__ __launch_bounds__(256)
void gemm_bf16_kernel(const float* __restrict__ A, const __bf16* __restrict__ Wp,
                      const float* __restrict__ bin, const float* __restrict__ bout,
                      float* __restrict__ Out, int nrows) {
    const int lane = threadIdx.x & 31;
    const int wave = threadIdx.x >> 5;
    const int nstrips = nrows >> 4;
    const int strip = blockIdx.x * (blockDim.x >> 5) + wave;   // wave-uniform
    if (strip >= nstrips) return;                              // whole wave exits: EXEC all-1s for WMMA
    const int row0 = strip << 4;
    const int half = lane >> 4;
    const int l15  = lane & 15;
    constexpr int NT = NOUT / 16;

    v8f acc[NT] = {};
    const float* arow = A + (size_t)(row0 + l15) * K;
    const v16bf* wfrag = (const v16bf*)Wp;

    for (int k0 = 0, kt = 0; k0 < K; k0 += 32, ++kt) {
        // A fragment: two 8-float runs per half, each as 2x float4 (32B aligned)
        const float* ap = arow + k0 + half * 8;
        float4 a0 = *(const float4*)(ap);
        float4 a1 = *(const float4*)(ap + 4);
        float4 a2 = *(const float4*)(ap + 16);
        float4 a3 = *(const float4*)(ap + 20);
        if constexpr (FUSE_IN) {
            const float* bp = bin + k0 + half * 8;
            a0 = relu_bias4(a0, *(const float4*)(bp));
            a1 = relu_bias4(a1, *(const float4*)(bp + 4));
            a2 = relu_bias4(a2, *(const float4*)(bp + 16));
            a3 = relu_bias4(a3, *(const float4*)(bp + 20));
        }
        v16bf afrag;
        afrag[0]  = (__bf16)a0.x; afrag[1]  = (__bf16)a0.y;
        afrag[2]  = (__bf16)a0.z; afrag[3]  = (__bf16)a0.w;
        afrag[4]  = (__bf16)a1.x; afrag[5]  = (__bf16)a1.y;
        afrag[6]  = (__bf16)a1.z; afrag[7]  = (__bf16)a1.w;
        afrag[8]  = (__bf16)a2.x; afrag[9]  = (__bf16)a2.y;
        afrag[10] = (__bf16)a2.z; afrag[11] = (__bf16)a2.w;
        afrag[12] = (__bf16)a3.x; afrag[13] = (__bf16)a3.y;
        afrag[14] = (__bf16)a3.z; afrag[15] = (__bf16)a3.w;

#pragma unroll
        for (int t = 0; t < NT; ++t) {
            const v16bf bfrag = wfrag[((size_t)(kt * NT + t) * 32) + lane];  // one 32B coalesced load
            acc[t] = __builtin_amdgcn_wmma_f32_16x16x32_bf16(
                false, afrag, false, bfrag, (short)0, acc[t], false, false);
        }
    }

#pragma unroll
    for (int t = 0; t < NT; ++t) {
        const int col = t * 16 + l15;
        float bo = 0.0f;
        if constexpr (BIAS_OUT) bo = bout[col];
#pragma unroll
        for (int j = 0; j < 8; ++j) {
            const int r = row0 + half * 8 + j;                // C/D: VGPR j -> row j + 8*half
            Out[(size_t)r * NOUT + col] = acc[t][j] + bo;
        }
    }
}

// ---------------- message passing: Acc[dst] += H[src] * dinv[src]*dinv[dst] ----------------

template <int D>
__global__ __launch_bounds__(256)
void agg_edges_kernel(const int* __restrict__ ei, const float* __restrict__ dinv,
                      const float* __restrict__ H, float* __restrict__ Acc, int nE) {
    constexpr int CH = D / 4;
    const long long tid = (long long)blockIdx.x * blockDim.x + threadIdx.x;
    const int e = (int)(tid / CH);
    const int c = (int)(tid % CH);
    if (e >= nE) return;
    const int s = ei[e];
    const int d = ei[nE + e];
    const float w = dinv[s] * dinv[d];
    const float4 v = *(const float4*)(H + (size_t)s * D + c * 4);
    float* o = Acc + (size_t)d * D + c * 4;
    atomicAdd(o + 0, v.x * w);
    atomicAdd(o + 1, v.y * w);
    atomicAdd(o + 2, v.z * w);
    atomicAdd(o + 3, v.w * w);
}

// self-loop term, run after the edge pass (stream-ordered, so plain RMW is safe)
template <int D>
__global__ __launch_bounds__(256)
void agg_self_kernel(const float* __restrict__ dinv, const float* __restrict__ H,
                     float* __restrict__ Acc, int n) {
    constexpr int CH = D / 4;
    const long long tid = (long long)blockIdx.x * blockDim.x + threadIdx.x;
    const int i = (int)(tid / CH);
    const int c = (int)(tid % CH);
    if (i >= n) return;
    const float w = dinv[i] * dinv[i];
    const float4 v = *(const float4*)(H + (size_t)i * D + c * 4);
    float4 a = *(const float4*)(Acc + (size_t)i * D + c * 4);
    a.x += v.x * w; a.y += v.y * w; a.z += v.z * w; a.w += v.w * w;
    *(float4*)(Acc + (size_t)i * D + c * 4) = a;
}

// ---------------- launcher ----------------

extern "C" void kernel_launch(void* const* d_in, const int* in_sizes, int n_in,
                              void* d_out, int out_size, void* d_ws, size_t ws_size,
                              hipStream_t stream) {
    const float* x   = (const float*)d_in[0];
    const int*   ei  = (const int*)  d_in[1];   // [2,E] src row then dst row
    const float* W1  = (const float*)d_in[2];
    const float* b1  = (const float*)d_in[3];
    const float* W2  = (const float*)d_in[4];
    const float* b2  = (const float*)d_in[5];
    const float* fcW = (const float*)d_in[6];
    const float* fcb = (const float*)d_in[7];
    float* out = (float*)d_out;

    const int N = in_sizes[0] / D_IN;
    const int E = in_sizes[1] / 2;

    // workspace layout:
    //   bf16 packed weights: W1p 16384 | W2p 8192 | fcWp 4096  (= 28672 bf16 = 14336 floats)
    //   deg/dinv [N] | buf1 [N*128] | buf2 [N*128]
    float*  ws   = (float*)d_ws;
    __bf16* W1p  = (__bf16*)ws;
    __bf16* W2p  = W1p + D_IN * D_HID;
    __bf16* fcWp = W2p + D_HID * D_OUT;
    float*  deg  = ws + 14336;
    float*  buf1 = deg + (((size_t)N + 255) & ~(size_t)255);
    float*  buf2 = buf1 + (size_t)N * D_IN;

    const int TB = 256;
    const int wpb = TB / 32;
    const int strips = N / 16;                       // 50000/16 = 3125 exactly
    dim3 gemm_grid((strips + wpb - 1) / wpb);

    // --- pack weights to bf16 fragment order ---
    pack_w_kernel<D_IN,  D_HID><<<(D_IN * D_HID + TB - 1) / TB, TB, 0, stream>>>(W1, W1p);
    pack_w_kernel<D_HID, D_OUT><<<(D_HID * D_OUT + TB - 1) / TB, TB, 0, stream>>>(W2, W2p);
    pack_w_kernel<D_OUT, D_OUT><<<(D_OUT * D_OUT + TB - 1) / TB, TB, 0, stream>>>(fcW, fcWp);

    // --- symmetric GCN normalization ---
    init_deg_kernel<<<(N + TB - 1) / TB, TB, 0, stream>>>(deg, N);
    count_deg_kernel<<<(E + TB - 1) / TB, TB, 0, stream>>>(ei, deg, E);
    rsqrt_kernel<<<(N + TB - 1) / TB, TB, 0, stream>>>(deg, N);   // deg -> dinv in place

    // --- layer 1: H1 = X @ W1 ; A1 = segsum(norm * H1[src]) ---
    gemm_bf16_kernel<D_IN, D_HID, false, false>
        <<<gemm_grid, TB, 0, stream>>>(x, W1p, nullptr, nullptr, buf1, N);
    {
        long long nz = (long long)N * D_HID;
        zero_kernel<<<(unsigned)((nz + TB - 1) / TB), TB, 0, stream>>>(buf2, nz);
        long long tot = (long long)E * (D_HID / 4);
        agg_edges_kernel<D_HID><<<(unsigned)((tot + TB - 1) / TB), TB, 0, stream>>>(ei, deg, buf1, buf2, E);
        long long ts = (long long)N * (D_HID / 4);
        agg_self_kernel<D_HID><<<(unsigned)((ts + TB - 1) / TB), TB, 0, stream>>>(deg, buf1, buf2, N);
    }

    // --- layer 2: H2 = relu(A1 + b1) @ W2 ; A2 = segsum(norm * H2[src]) ---
    gemm_bf16_kernel<D_HID, D_OUT, true, false>
        <<<gemm_grid, TB, 0, stream>>>(buf2, W2p, b1, nullptr, buf1, N);
    {
        long long nz = (long long)N * D_OUT;
        zero_kernel<<<(unsigned)((nz + TB - 1) / TB), TB, 0, stream>>>(buf2, nz);
        long long tot = (long long)E * (D_OUT / 4);
        agg_edges_kernel<D_OUT><<<(unsigned)((tot + TB - 1) / TB), TB, 0, stream>>>(ei, deg, buf1, buf2, E);
        long long ts = (long long)N * (D_OUT / 4);
        agg_self_kernel<D_OUT><<<(unsigned)((ts + TB - 1) / TB), TB, 0, stream>>>(deg, buf1, buf2, N);
    }

    // --- head: out = relu(A2 + b2) @ fc_W + fc_b ---
    gemm_bf16_kernel<D_OUT, D_OUT, true, true>
        <<<gemm_grid, TB, 0, stream>>>(buf2, fcWp, b2, fcb, out, N);
}
